// EMAVectorQuantizer_13984413516274
// MI455X (gfx1250) — compile-verified, matrix operand order checked by reading, not run on hardware
//
#include <hip/hip_runtime.h>
#include <hip/hip_bf16.h>

typedef __attribute__((ext_vector_type(16))) __bf16 v16bf;
typedef __attribute__((ext_vector_type(8)))  float  v8f;
typedef __attribute__((address_space(3)))    __bf16 lds_bf16;

#define NTOK   8192      // tokens (8*32*32)
#define NCODE  8192      // codebook entries
#define DIM    256       // feature dim
#define NRB    512       // 16-row blocks of tokens
#define NNB    512       // 16-col blocks of codes

__device__ __forceinline__ void atomicAddF(float* p, float v) {
    __hip_atomic_fetch_add(p, v, __ATOMIC_RELAXED, __HIP_MEMORY_SCOPE_AGENT);
}

// K index inside a 32-wide chunk for fragment element e, lane l (ISA 7.12.2, 16-bit A 16x32)
__device__ __forceinline__ int frag_k(int e, int lane) {
    return (e & 7) + 8 * (2 * (e >> 3) + (lane >> 4));
}

// async stage 32 bytes/thread from global into LDS (2x B128, ASYNCcnt += 2)
__device__ __forceinline__ void async_copy32(unsigned lds_byte_addr, const char* gptr) {
    unsigned long long g = (unsigned long long)(size_t)gptr;
    asm volatile("global_load_async_to_lds_b128 %0, %1, off\n\t"
                 "global_load_async_to_lds_b128 %0, %1, off offset:16"
                 :: "v"(lds_byte_addr), "v"(g) : "memory");
}

// ---- pack tokens (NCHW f32 -> bf16 WMMA-A fragment order) -------------------
__global__ __launch_bounds__(256) void pack_a(const float* __restrict__ z,
                                              __bf16* __restrict__ Abuf) {
    int t = blockIdx.x * 256 + threadIdx.x;           // 2,097,152 elements
    int e = t & 15, lane = (t >> 4) & 31, c = (t >> 9) & 7, rb = t >> 12;
    int d = c * 32 + frag_k(e, lane);
    int n = rb * 16 + (lane & 15);
    int b = n >> 10, hw = n & 1023;
    Abuf[t] = (__bf16)z[((size_t)(b * DIM + d) << 10) + hw];
}

// ---- pack codebook (f32 [D,K] -> bf16 WMMA-B fragment order) ----------------
__global__ __launch_bounds__(256) void pack_b(const float* __restrict__ weight,
                                              __bf16* __restrict__ Bbuf) {
    int t = blockIdx.x * 256 + threadIdx.x;           // 2,097,152 elements
    int e = t & 15, lane = (t >> 4) & 31, c = (t >> 9) & 7, nb = t >> 12;
    int d = c * 32 + frag_k(e, lane);
    int k = nb * 16 + (lane & 15);
    Bbuf[t] = (__bf16)weight[(size_t)d * NCODE + k];
}

// ---- exact f32 ||w_k||^2 (coalesced across k) -------------------------------
__global__ __launch_bounds__(256) void calc_w2(const float* __restrict__ weight,
                                               float* __restrict__ w2) {
    int k = blockIdx.x * 256 + threadIdx.x;
    float s = 0.f;
    for (int d = 0; d < DIM; ++d) { float v = weight[(size_t)d * NCODE + k]; s += v * v; }
    w2[k] = s;
}

// ---- fused GEMM (bf16 WMMA, f32 acc) + row argmin ---------------------------
// 8 waves/block share B tiles via async global->LDS, 3-deep pipeline.
__global__ __launch_bounds__(256) void gemm_argmin(const __bf16* __restrict__ Abuf,
                                                   const __bf16* __restrict__ Bbuf,
                                                   const float* __restrict__ w2,
                                                   int* __restrict__ idxout) {
    __shared__ __attribute__((aligned(32))) __bf16 tile[3][4096];   // 3 x 8 KB

    const int lane = threadIdx.x & 31;
    const int wave = threadIdx.x >> 5;
    const int rb   = blockIdx.x * 8 + wave;           // 64 blocks * 8 waves = 512

    // LDS byte address of this thread's 32 B slice within slot 0
    const unsigned ldsbase =
        (unsigned)(size_t)(lds_bf16*)&tile[0][0] + (unsigned)threadIdx.x * 32u;
    const char* gbase = (const char*)Bbuf + (size_t)threadIdx.x * 32u;

    // A fragments for this 16-row block (64 VGPRs)
    v16bf a[8];
    const __bf16* Ap = Abuf + (size_t)rb * 4096 + lane * 16;
#pragma unroll
    for (int c = 0; c < 8; ++c) a[c] = *(const v16bf*)(Ap + c * 512);

    float best[8]; int bidx[8];
#pragma unroll
    for (int v = 0; v < 8; ++v) { best[v] = 3.4e38f; bidx[v] = 0; }

    // prologue: stage tiles 0 and 1
    async_copy32(ldsbase,         gbase);
    async_copy32(ldsbase + 8192u, gbase + 8192);

    const int col = lane & 15;
    int slot = 0;
    for (int nb = 0; nb < NNB; ++nb) {
        // own tile-nb loads done (tile nb+1 may still fly); barrier -> all waves' slices in
        if (nb == NNB - 1) asm volatile("s_wait_asynccnt 0x0" ::: "memory");
        else               asm volatile("s_wait_asynccnt 0x2" ::: "memory");
        __syncthreads();

        if (nb + 2 < NNB) {                 // stage tile nb+2 into slot read last iter
            int s2 = slot + 2; if (s2 >= 3) s2 -= 3;
            async_copy32(ldsbase + (unsigned)s2 * 8192u, gbase + (size_t)(nb + 2) * 8192u);
        }

        const __bf16* Bp = &tile[slot][lane * 16];
        v8f acc0 = {}, acc1 = {};
#pragma unroll
        for (int c = 0; c < 8; c += 2) {
            v16bf b0 = *(const v16bf*)(Bp + c * 512);
            v16bf b1 = *(const v16bf*)(Bp + (c + 1) * 512);
            acc0 = __builtin_amdgcn_wmma_f32_16x16x32_bf16(false, a[c],     false, b0,
                                                           (short)0, acc0, false, false);
            acc1 = __builtin_amdgcn_wmma_f32_16x16x32_bf16(false, a[c + 1], false, b1,
                                                           (short)0, acc1, false, false);
        }
        const int   code = nb * 16 + col;
        const float wc   = w2[code];
#pragma unroll
        for (int v = 0; v < 8; ++v) {
            float dist = wc - 2.0f * (acc0[v] + acc1[v]);   // ||z||^2 dropped: row-constant
            if (dist < best[v]) { best[v] = dist; bidx[v] = code; }
        }
        ++slot; if (slot == 3) slot = 0;
    }

    // min+argmin across the 16 lanes holding each row (D layout: lane = {half, N})
#pragma unroll
    for (int v = 0; v < 8; ++v) {
        float d = best[v]; int i = bidx[v];
#pragma unroll
        for (int m = 1; m < 16; m <<= 1) {
            float od = __shfl_xor(d, m, 32);
            int   oi = __shfl_xor(i, m, 32);
            if (od < d || (od == d && oi < i)) { d = od; i = oi; }
        }
        if ((lane & 15) == 0) idxout[rb * 16 + (v + 8 * (lane >> 4))] = i;
    }
}

// ---- counts ------------------------------------------------------------------
__global__ __launch_bounds__(256) void count_kernel(const int* __restrict__ idx,
                                                    float* __restrict__ counts) {
    int n = blockIdx.x * 256 + threadIdx.x;
    atomicAddF(&counts[idx[n]], 1.0f);
}

// ---- fused: z_q_out (NCHW), embed_sum scatter, loss partial ------------------
__global__ __launch_bounds__(256) void scatter_kernel(const float* __restrict__ z,
                                                      const float* __restrict__ weight,
                                                      const int* __restrict__ idx,
                                                      float* __restrict__ embed_sum,
                                                      float* __restrict__ zq_out,
                                                      float* __restrict__ lossAcc) {
    const int d = blockIdx.y;                                   // 0..255
    const int n = blockIdx.x * 256 + threadIdx.x;               // 0..8191
    const int b = n >> 10, hw = n & 1023;
    const size_t zoff = ((size_t)(b * DIM + d) << 10) + hw;     // coalesced in hw
    const float zv = z[zoff];
    const int   code = idx[n];
    const float wv = weight[(size_t)d * NCODE + code];
    zq_out[zoff] = wv;                                          // straight-through == z_q
    atomicAddF(&embed_sum[(size_t)d * NCODE + code], zv);
    const float diff = wv - zv;
    __shared__ float red[256];
    red[threadIdx.x] = diff * diff;
    __syncthreads();
    for (int s = 128; s > 0; s >>= 1) {
        if (threadIdx.x < s) red[threadIdx.x] += red[threadIdx.x + s];
        __syncthreads();
    }
    if (threadIdx.x == 0) atomicAddF(lossAcc, red[0]);
}

// ---- n = sum(new_cluster_size), perplexity ----------------------------------
__global__ __launch_bounds__(1024) void stats_kernel(const float* __restrict__ counts,
                                                     const float* __restrict__ cluster_size,
                                                     float* __restrict__ scalars,
                                                     float* __restrict__ out_perp) {
    const int t = threadIdx.x;
    float sn = 0.f, sp = 0.f;
    for (int k = t; k < NCODE; k += 1024) {
        sn += 0.99f * cluster_size[k] + 0.01f * counts[k];
        float p = counts[k] * (1.0f / (float)NTOK);
        sp += p * logf(p + 1e-10f);
    }
    __shared__ float rn[1024], rp[1024];
    rn[t] = sn; rp[t] = sp;
    __syncthreads();
    for (int s = 512; s > 0; s >>= 1) {
        if (t < s) { rn[t] += rn[t + s]; rp[t] += rp[t + s]; }
        __syncthreads();
    }
    if (t == 0) { scalars[0] = rn[0]; out_perp[0] = expf(-rp[0]); }
}

// ---- new_cluster_size, idx (as float), loss ---------------------------------
__global__ __launch_bounds__(256) void small_out_kernel(const float* __restrict__ cluster_size,
                                                        const float* __restrict__ counts,
                                                        const int* __restrict__ idx,
                                                        const float* __restrict__ scalars,
                                                        float* __restrict__ out_ncs,
                                                        float* __restrict__ out_idx,
                                                        float* __restrict__ out_loss) {
    int k = blockIdx.x * 256 + threadIdx.x;
    out_ncs[k] = 0.99f * cluster_size[k] + 0.01f * counts[k];
    out_idx[k] = (float)idx[k];
    if (k == 0) out_loss[0] = 0.25f * scalars[1] / 2097152.0f;
}

// ---- new_embed_avg, new_weight ----------------------------------------------
__global__ __launch_bounds__(256) void codebook_kernel(const float* __restrict__ embed_avg,
                                                       const float* __restrict__ embed_sum,
                                                       const float* __restrict__ cluster_size,
                                                       const float* __restrict__ counts,
                                                       const float* __restrict__ scalars,
                                                       float* __restrict__ out_w,
                                                       float* __restrict__ out_ea) {
    size_t t = (size_t)blockIdx.x * 256 + threadIdx.x;          // 2,097,152
    int k = (int)(t & (NCODE - 1));
    float nea = 0.99f * embed_avg[t] + 0.01f * embed_sum[t];
    out_ea[t] = nea;
    float n   = scalars[0];
    float ncs = 0.99f * cluster_size[k] + 0.01f * counts[k];
    float cs  = (ncs + 1e-5f) / (n + (float)NCODE * 1e-5f) * n;
    out_w[t] = nea / cs;
}

extern "C" void kernel_launch(void* const* d_in, const int* in_sizes, int n_in,
                              void* d_out, int out_size, void* d_ws, size_t ws_size,
                              hipStream_t stream) {
    const float* z            = (const float*)d_in[0];
    const float* weight       = (const float*)d_in[1];
    const float* cluster_size = (const float*)d_in[2];
    const float* embed_avg    = (const float*)d_in[3];
    float* out = (float*)d_out;

    char* ws = (char*)d_ws;
    __bf16* Abuf    = (__bf16*)ws;                                   // 4 MB
    __bf16* Bbuf    = (__bf16*)(ws + (4u << 20));                    // 4 MB
    float* w2       = (float*)(ws + (8u << 20));                     // 32 KB
    int*   idxbuf   = (int*)  (ws + (8u << 20) + (32u << 10));       // 32 KB
    float* counts   = (float*)(ws + (8u << 20) + (64u << 10));       // 32 KB
    float* embedsum = (float*)(ws + (8u << 20) + (96u << 10));       // 8 MB
    float* scalars  = (float*)(ws + (16u << 20) + (96u << 10));      // few B

    // zero counts + embed_sum + scalars (contiguous region)
    hipMemsetAsync(counts, 0, (32u << 10) + (8u << 20) + 256, stream);

    pack_a  <<<8192, 256, 0, stream>>>(z, Abuf);
    pack_b  <<<8192, 256, 0, stream>>>(weight, Bbuf);
    calc_w2 <<<32,   256, 0, stream>>>(weight, w2);

    gemm_argmin<<<64, 256, 0, stream>>>(Abuf, Bbuf, w2, idxbuf);

    count_kernel<<<32, 256, 0, stream>>>(idxbuf, counts);
    scatter_kernel<<<dim3(32, 256), 256, 0, stream>>>(z, weight, idxbuf, embedsum,
                                                      out /* z_q_out at 0 */, scalars + 1);
    stats_kernel<<<1, 1024, 0, stream>>>(counts, cluster_size, scalars, out + 2097153);

    small_out_kernel<<<32, 256, 0, stream>>>(cluster_size, counts, idxbuf, scalars,
                                             out + 4202498,   // new_cluster_size
                                             out + 2097154,   // idx (as float)
                                             out + 2097152);  // loss
    codebook_kernel<<<8192, 256, 0, stream>>>(embed_avg, embedsum, cluster_size, counts,
                                              scalars,
                                              out + 2105346,   // new_weight
                                              out + 4210690);  // new_embed_avg
}
